// MultiAgents_83210696392760
// MI455X (gfx1250) — matrix operand order repaired and checked by reading, not compiled, not run on hardware
//
#include <hip/hip_runtime.h>

#define BATCH 16384
#define IN    784
#define H     256
#define NA    4
#define NO    10
#define ND    32

typedef __attribute__((ext_vector_type(16))) __bf16       v16bf;
typedef __attribute__((ext_vector_type(8)))  float        v8f;
typedef __attribute__((ext_vector_type(4)))  unsigned int u32x4;

struct U32x8 { u32x4 lo, hi; };

static __device__ __forceinline__ unsigned short f2bf(float f) {
    unsigned int u = __float_as_uint(f);
    unsigned int r = u + 0x7FFFu + ((u >> 16) & 1u);   // round-to-nearest-even
    return (unsigned short)(r >> 16);
}

static __device__ __forceinline__ v16bf make_frag(u32x4 lo, u32x4 hi) {
    U32x8 t; t.lo = lo; t.hi = hi;
    return __builtin_bit_cast(v16bf, t);
}

// CDNA5 async global->LDS copy (16B per lane), tracked by ASYNCcnt.
// Flat LDS addresses truncate to the LDS byte offset (ISA 10.2), so the low
// 32 bits of the generic shared pointer are a valid VDST LDS address.
static __device__ __forceinline__ void async_b128(void* lds_dst, const void* gsrc) {
    unsigned loff = (unsigned)(size_t)lds_dst;
    asm volatile("global_load_async_to_lds_b128 %0, %1, off"
                 :: "v"(loff), "v"(gsrc) : "memory");
}
static __device__ __forceinline__ void wait_async0() {
    asm volatile("s_wait_asynccnt 0x0" ::: "memory");
}

// ---------------- prep: fp32 -> bf16 conversions ----------------
__global__ void k_cvt(const float* __restrict__ src, unsigned short* __restrict__ dst, int n) {
    int i = blockIdx.x * 256 + threadIdx.x;
    if (i < n) dst[i] = f2bf(src[i]);
}

// W1 (a, i=IN, o=H) -> w1t (a, o, i)  [B^T layout for GEMM1]
__global__ void k_tr_w1(const float* __restrict__ W1, unsigned short* __restrict__ w1t) {
    int i = blockIdx.x * 256 + threadIdx.x;
    if (i >= NA * IN * H) return;
    int a = i / (IN * H);
    int rem = i - a * IN * H;
    int k = rem / H;
    int o = rem - k * H;
    w1t[((size_t)a * H + o) * IN + k] = f2bf(W1[i]);
}

// ---------------- kernel 1: u = squash(x @ W1[a] + b1[a]) ----------------
// grid (BATCH/32, NA), 256 threads (8 waves). M=32, N=256, K=784 (64-chunks).
__global__ __launch_bounds__(256) void k_u_gemm(
    const unsigned short* __restrict__ xb,   // (BATCH, IN) bf16
    const unsigned short* __restrict__ w1t,  // (NA, H, IN) bf16
    const float*          __restrict__ b1,   // (NA, H)
    unsigned short*       __restrict__ ub)   // (BATCH, NA*H) bf16
{
    __shared__ __align__(16) unsigned short smem[32 * 72 + 256 * 72]; // 41472 B
    __shared__ float spart[32][8];
    __shared__ float sscale[32];

    const int tid  = threadIdx.x;
    const int wave = tid >> 5, lane = tid & 31;
    const int rb   = blockIdx.x * 32;
    const int a    = blockIdx.y;

    unsigned short* As = smem;            // [32][72]
    unsigned short* Bs = smem + 32 * 72;  // [256][72]

    const unsigned short* xrow = xb  + (size_t)rb * IN;
    const unsigned short* wrow = w1t + (size_t)a * H * IN;

    // per-thread staging coordinates (16B chunks)
    const int rA  = tid >> 3;          // A row 0..31
    const int c8  = (tid & 7) * 8;     // k-offset within chunk

    v8f acc[2][2] = {};

    for (int kc = 0; kc < IN; kc += 64) {
        if (kc + 64 <= IN) {
            // fast path: fully-valid 64-wide chunk, straight-line async copies
            async_b128(As + rA * 72 + c8, xrow + (size_t)rA * IN + kc + c8);
            #pragma unroll
            for (int j = 0; j < 8; ++j) {
                int idx = tid + 256 * j;
                int n = idx >> 3, cc = (idx & 7) * 8;
                async_b128(Bs + n * 72 + cc, wrow + (size_t)n * IN + kc + cc);
            }
        } else {
            // tail chunk (once): k = 768..783 valid, rest zero-filled
            if (c8 < 16) {
                async_b128(As + rA * 72 + c8, xrow + (size_t)rA * IN + kc + c8);
            } else {
                u32x4 z = {};
                *(u32x4*)(As + rA * 72 + c8) = z;
            }
            #pragma unroll
            for (int j = 0; j < 8; ++j) {
                int idx = tid + 256 * j;
                int n = idx >> 3, cc = (idx & 7) * 8;
                if (cc < 16) {
                    async_b128(Bs + n * 72 + cc, wrow + (size_t)n * IN + kc + cc);
                } else {
                    u32x4 z = {};
                    *(u32x4*)(Bs + n * 72 + cc) = z;
                }
            }
        }
        wait_async0();
        __syncthreads();

        #pragma unroll
        for (int ks = 0; ks < 64; ks += 32) {
            v16bf af[2];
            #pragma unroll
            for (int mt = 0; mt < 2; ++mt) {
                int row = mt * 16 + (lane & 15);
                int kA  = ks + ((lane >> 4) << 3);
                af[mt] = make_frag(*(const u32x4*)(As + row * 72 + kA),
                                   *(const u32x4*)(As + row * 72 + kA + 16));
            }
            v16bf bfr[2];
            #pragma unroll
            for (int j = 0; j < 2; ++j) {
                int n  = (wave * 2 + j) * 16 + (lane & 15);
                int kB = ks + ((lane >> 4) << 3);
                bfr[j] = make_frag(*(const u32x4*)(Bs + n * 72 + kB),
                                   *(const u32x4*)(Bs + n * 72 + kB + 16));
            }
            #pragma unroll
            for (int mt = 0; mt < 2; ++mt)
                #pragma unroll
                for (int j = 0; j < 2; ++j)
                    acc[mt][j] = __builtin_amdgcn_wmma_f32_16x16x32_bf16(
                        false, af[mt], false, bfr[j], (short)0, acc[mt][j], false, false);
        }
        __syncthreads();
    }

    // spill raw + bias to LDS (reuse smem as fp32 Us[32][256])
    float* Us = (float*)smem;
    #pragma unroll
    for (int mt = 0; mt < 2; ++mt)
        #pragma unroll
        for (int j = 0; j < 2; ++j) {
            int col = (wave * 2 + j) * 16 + (lane & 15);
            float bias = b1[a * H + col];
            #pragma unroll
            for (int i = 0; i < 8; ++i) {
                int row = mt * 16 + i + ((lane >> 4) << 3);
                Us[row * 256 + col] = acc[mt][j][i] + bias;
            }
        }
    __syncthreads();

    { // squash row norms
        int r = tid >> 3, seg = tid & 7;
        float s = 0.f;
        #pragma unroll
        for (int c = 0; c < 32; ++c) { float v = Us[r * 256 + seg * 32 + c]; s += v * v; }
        spart[r][seg] = s;
    }
    __syncthreads();
    if (tid < 32) {
        float sn = 0.f;
        #pragma unroll
        for (int j = 0; j < 8; ++j) sn += spart[tid][j];
        sscale[tid] = sn / (1.f + sn) * rsqrtf(sn + 1e-8f);
    }
    __syncthreads();
    {
        int r = tid >> 3, cb = (tid & 7) * 32;
        float sc = sscale[r];
        unsigned short* dst = ub + (size_t)(rb + r) * (NA * H) + a * H + cb;
        for (int c = 0; c < 32; ++c) dst[c] = f2bf(Us[r * 256 + cb + c] * sc);
    }
}

// ---------------- kernel 2: pred GEMM + 3-iter routing + output norm ----------------
// grid BATCH/8, 256 threads (8 waves). 8 batch rows per block.
__global__ __launch_bounds__(256) void k_pred_route(
    const unsigned short* __restrict__ ub,  // (BATCH, NA*H) bf16
    const unsigned short* __restrict__ wb,  // (NA, NO*ND, H) bf16  (= B^T)
    float*                __restrict__ out) // (BATCH, NO)
{
    __shared__ __align__(16) unsigned short uS[16 * 264]; // 8448 B
    __shared__ float predS[8 * 1280];                     // 40960 B

    const int tid  = threadIdx.x;
    const int wave = tid >> 5, lane = tid & 31;
    const int rb   = blockIdx.x * 8;

    // rows 8..15 are static zero padding for the 16-row WMMA A-tile: fill once
    {
        int idx = tid + 256;
        int row = idx >> 5, c8 = (idx & 31) * 8;
        u32x4 z = {};
        *(u32x4*)(uS + row * 264 + c8) = z;
    }

    for (int a = 0; a < NA; ++a) {
        __syncthreads();   // previous agent's tile fully consumed
        {   // stage rows 0..7: one unguarded async 16B chunk per thread
            int row = tid >> 5, c8 = (tid & 31) * 8;
            async_b128(uS + row * 264 + c8,
                       ub + (size_t)(rb + row) * (NA * H) + a * H + c8);
        }
        wait_async0();
        __syncthreads();

        for (int nt = wave; nt < 20; nt += 8) {
            v8f acc = {};
            #pragma unroll
            for (int ks = 0; ks < 8; ++ks) {
                int row = lane & 15;
                int kA  = ks * 32 + ((lane >> 4) << 3);
                v16bf af = make_frag(*(const u32x4*)(uS + row * 264 + kA),
                                     *(const u32x4*)(uS + row * 264 + kA + 16));
                int n = nt * 16 + (lane & 15);
                const unsigned short* bp =
                    wb + ((size_t)(a * NO * ND + n)) * H + ks * 32 + ((lane >> 4) << 3);
                v16bf bfr = make_frag(*(const u32x4*)(bp), *(const u32x4*)(bp + 16));
                acc = __builtin_amdgcn_wmma_f32_16x16x32_bf16(
                    false, af, false, bfr, (short)0, acc, false, false);
            }
            if ((lane >> 4) == 0) {
                int col = a * NO * ND + nt * 16 + (lane & 15);
                #pragma unroll
                for (int i = 0; i < 8; ++i) predS[i * 1280 + col] = acc[i];
            }
        }
    }
    __syncthreads();

    // routing by agreement: wave w handles row w, lane = capsule dim d
    const int r = wave;
    float predv[NA][NO];
    #pragma unroll
    for (int a = 0; a < NA; ++a)
        #pragma unroll
        for (int o = 0; o < NO; ++o)
            predv[a][o] = predS[r * 1280 + a * NO * ND + o * ND + lane];

    float logits[NA][NO] = {};
    float vnorm[NO];
    #pragma unroll
    for (int it = 0; it < 3; ++it) {
        float cc[NA][NO];
        #pragma unroll
        for (int a = 0; a < NA; ++a) {
            float m = logits[a][0];
            #pragma unroll
            for (int o = 1; o < NO; ++o) m = fmaxf(m, logits[a][o]);
            float sum = 0.f;
            #pragma unroll
            for (int o = 0; o < NO; ++o) { cc[a][o] = __expf(logits[a][o] - m); sum += cc[a][o]; }
            float inv = 1.f / sum;
            #pragma unroll
            for (int o = 0; o < NO; ++o) cc[a][o] *= inv;
        }
        float vv[NO];
        #pragma unroll
        for (int o = 0; o < NO; ++o) {
            float s = 0.f;
            #pragma unroll
            for (int a = 0; a < NA; ++a) s += cc[a][o] * predv[a][o];
            float t = s * s;
            #pragma unroll
            for (int off = 16; off > 0; off >>= 1) t += __shfl_xor(t, off, 32);
            float sn = t;
            float scale = sn / (1.f + sn) * rsqrtf(sn + 1e-8f);
            vv[o]    = s * scale;
            vnorm[o] = sqrtf(sn) * scale;   // ||squash(s)||
        }
        if (it < 2) {
            #pragma unroll
            for (int a = 0; a < NA; ++a)
                #pragma unroll
                for (int o = 0; o < NO; ++o) {
                    float t = predv[a][o] * vv[o];
                    #pragma unroll
                    for (int off = 16; off > 0; off >>= 1) t += __shfl_xor(t, off, 32);
                    logits[a][o] += t;
                }
        }
    }
    if (lane == 0) {
        float* op = out + (size_t)(rb + r) * NO;
        #pragma unroll
        for (int o = 0; o < NO; ++o) op[o] = vnorm[o];
    }
}

extern "C" void kernel_launch(void* const* d_in, const int* in_sizes, int n_in,
                              void* d_out, int out_size, void* d_ws, size_t ws_size,
                              hipStream_t stream) {
    const float* x  = (const float*)d_in[0];
    const float* W1 = (const float*)d_in[1];
    const float* b1 = (const float*)d_in[2];
    const float* W  = (const float*)d_in[3];
    float* out = (float*)d_out;

    unsigned char* ws = (unsigned char*)d_ws;
    size_t off_xb  = 0;
    size_t off_w1t = off_xb  + (size_t)BATCH * IN * 2;        // 25,690,112
    size_t off_wb  = off_w1t + (size_t)NA * H * IN * 2;       // +1,605,632
    size_t off_ub  = off_wb  + (size_t)NA * NO * ND * H * 2;  // +655,360
    unsigned short* xb  = (unsigned short*)(ws + off_xb);
    unsigned short* w1t = (unsigned short*)(ws + off_w1t);
    unsigned short* wb  = (unsigned short*)(ws + off_wb);
    unsigned short* ub  = (unsigned short*)(ws + off_ub);

    int nx = BATCH * IN;
    k_cvt<<<(nx + 255) / 256, 256, 0, stream>>>(x, xb, nx);
    int nw1 = NA * IN * H;
    k_tr_w1<<<(nw1 + 255) / 256, 256, 0, stream>>>(W1, w1t);
    int nw = NA * NO * ND * H;
    k_cvt<<<(nw + 255) / 256, 256, 0, stream>>>(W, wb, nw);

    dim3 g1(BATCH / 32, NA);
    k_u_gemm<<<g1, 256, 0, stream>>>(xb, w1t, b1, ub);

    k_pred_route<<<BATCH / 8, 256, 0, stream>>>(ub, wb, out);
}
